// CrossAttentionBlock_22771916604084
// MI455X (gfx1250) — matrix operand order repaired and checked by reading, not compiled, not run on hardware
//
#include <hip/hip_runtime.h>

#define NVOX 13824      // 24*24*24
#define CCH  128
#define RCH  16
#define NT   (NVOX / 32)   // 432 key blocks of 32
#define VPITCH 40       // halfwords (80B row pitch, conflict-padded, 16B aligned)
#define PPITCH 40

typedef _Float16 v16h __attribute__((ext_vector_type(16)));
typedef float    v8f  __attribute__((ext_vector_type(8)));
typedef unsigned int v4u __attribute__((ext_vector_type(4)));
typedef int      v4i  __attribute__((ext_vector_type(4)));

union FragH { v16h v; v4u q[2]; };

#if defined(__has_builtin)
#if __has_builtin(__builtin_amdgcn_global_load_async_to_lds_b128)
#define USE_ASYNC 1
#endif
#endif
#ifndef USE_ASYNC
#define USE_ASYNC 0
#endif

#if USE_ASYNC
typedef __attribute__((address_space(1))) v4i GV4;   // global int4
typedef __attribute__((address_space(3))) v4i LV4;   // LDS int4

static __device__ __forceinline__ void async_copy16(const _Float16* g, _Float16* l) {
    __builtin_amdgcn_global_load_async_to_lds_b128(
        (GV4*)const_cast<_Float16*>(g), (LV4*)l, 0, 0);
}
#endif

// fold attention scale and log2(e) into q:  scores_log2 = (q.k) * 0.25 * log2(e)
#define QSCALE 0.3606737602222409f

// ---------------- projections: q = QSCALE*(Wq x_q + bq), k = Wk x_kv + bk ----------------
__global__ __launch_bounds__(256) void ca_proj_qk(
    const float* __restrict__ xq, const float* __restrict__ xkv,
    const float* __restrict__ Wq, const float* __restrict__ bq,
    const float* __restrict__ Wk, const float* __restrict__ bk,
    _Float16* __restrict__ Qh, _Float16* __restrict__ Kh)
{
    int tid   = threadIdx.x;
    int r     = tid & 15;
    int nl    = (tid >> 4) & 7;
    int which = tid >> 7;                 // 0 = q, 1 = k
    int n     = blockIdx.x * 8 + nl;
    const float* x = which ? xkv : xq;
    const float* W = which ? Wk  : Wq;
    const float* b = which ? bk  : bq;
    float acc = 0.0f;
    #pragma unroll 8
    for (int c = 0; c < CCH; ++c)
        acc = fmaf(W[r * CCH + c], x[(size_t)c * NVOX + n], acc);
    acc += b[r];
    if (which == 0) acc *= QSCALE;
    _Float16* dst = which ? Kh : Qh;
    dst[(size_t)n * RCH + r] = (_Float16)acc;
}

// ---------------- v = Wv x_kv + bv, stored channel-major f16 Vt[c][n] ----------------
__global__ __launch_bounds__(256) void ca_proj_v(
    const float* __restrict__ xkv, const float* __restrict__ Wv,
    const float* __restrict__ bv, _Float16* __restrict__ Vt)
{
    int idx = blockIdx.x * 256 + threadIdx.x;   // exactly C*N threads
    int n = idx % NVOX;
    int c = idx / NVOX;
    float acc = bv[c];
    #pragma unroll 8
    for (int cc = 0; cc < CCH; ++cc)
        acc = fmaf(Wv[c * CCH + cc], xkv[(size_t)cc * NVOX + n], acc);
    Vt[(size_t)c * NVOX + n] = (_Float16)acc;
}

// ---------------- fused flash attention: 64 queries per WG, 4 waves x 16 rows ----------------
__global__ __launch_bounds__(128) void ca_flash_attn(
    const _Float16* __restrict__ Qh,
    const _Float16* __restrict__ Kh,
    const _Float16* __restrict__ Vt,
    const float* __restrict__ xq,
    const float* __restrict__ gamma,
    float* __restrict__ out)
{
    __shared__ _Float16 ldsV[2][CCH * VPITCH];    // double-buffered V tile: 128ch x 32key
    __shared__ _Float16 ldsP[4 * 16 * PPITCH];    // per-wave P tile: 16q x 32key

    const int tid  = threadIdx.x;
    const int wave = tid >> 5;
    const int lane = tid & 31;
    const int l16  = lane & 15;
    const int hi   = lane >> 4;

    _Float16* pbuf = ldsP + wave * 16 * PPITCH;
    const int qbase = blockIdx.x * 64 + wave * 16;

    // staging-chunk coordinates for cooperative V loads (4 x 16B per thread)
    int vc[4], vkc[4];
    #pragma unroll
    for (int i = 0; i < 4; ++i) {
        int chunk = tid + i * 128;     // 0..511
        vc[i]  = chunk >> 2;           // channel row
        vkc[i] = chunk & 3;            // 8-key (16B) sub-chunk
    }

    // Q fragment: A-matrix 16x32 f16 layout, K dim padded 16->32 with zeros.
    FragH aq;
    const v4u zq = {0u, 0u, 0u, 0u};
    {
        const v4u* qp = (const v4u*)(Qh + (size_t)(qbase + l16) * RCH);
        aq.q[0] = hi ? zq : qp[0];   // lanes<16: r0..7 ; lanes>=16: 0 padding
        aq.q[1] = hi ? zq : qp[1];   // lanes<16: r8..15 ; lanes>=16: 0 padding
    }

    // all-ones B fragment (f16 1.0) for the running row-sum column
    FragH ones;
    {
        const v4u ov = {0x3C003C00u, 0x3C003C00u, 0x3C003C00u, 0x3C003C00u};
        ones.q[0] = ov; ones.q[1] = ov;
    }

    v8f O[9];                       // 8 channel tiles + 1 row-sum tile
    #pragma unroll
    for (int ct = 0; ct < 9; ++ct) { v8f z = {0,0,0,0,0,0,0,0}; O[ct] = z; }
    float m = -3.0e38f;

    // K fragment pipeline: lanes>=16 stay zero forever (r>=16 padding)
    FragH bk0, bk1;
    bk0.q[0] = zq; bk0.q[1] = zq; bk1.q[0] = zq; bk1.q[1] = zq;
    if (lane < 16) {
        const v4u* kp0 = (const v4u*)(Kh + (size_t)lane * RCH);
        bk0.q[0] = kp0[0]; bk0.q[1] = kp0[1];
        const v4u* kp1 = (const v4u*)(Kh + (size_t)(16 + lane) * RCH);
        bk1.q[0] = kp1[0]; bk1.q[1] = kp1[1];
    }

    // V prolog: tile 0 -> buffer 0
#if USE_ASYNC
    #pragma unroll
    for (int i = 0; i < 4; ++i)
        async_copy16(Vt + (size_t)vc[i] * NVOX + vkc[i] * 8,
                     &ldsV[0][vc[i] * VPITCH + vkc[i] * 8]);
#else
    #pragma unroll
    for (int i = 0; i < 4; ++i) {
        v4u d = *(const v4u*)(Vt + (size_t)vc[i] * NVOX + vkc[i] * 8);
        *(v4u*)(&ldsV[0][vc[i] * VPITCH + vkc[i] * 8]) = d;
    }
#endif

    int cur = 0;
    for (int it = 0; it < NT; ++it) {
#if USE_ASYNC
        asm volatile("s_wait_asynccnt 0x0" ::: "memory");  // tile for buf[cur] landed
#endif
        __syncthreads();   // buf[cur] ready for all; prior reads of buf[cur^1] done

        const int nit = (it + 1 == NT) ? 0 : it + 1;
        const int nk0 = nit * 32;

        // ---- start next V tile transfer (overlaps with compute below) ----
#if USE_ASYNC
        #pragma unroll
        for (int i = 0; i < 4; ++i)
            async_copy16(Vt + (size_t)vc[i] * NVOX + nk0 + vkc[i] * 8,
                         &ldsV[cur ^ 1][vc[i] * VPITCH + vkc[i] * 8]);
#else
        v4u vr0 = *(const v4u*)(Vt + (size_t)vc[0] * NVOX + nk0 + vkc[0] * 8);
        v4u vr1 = *(const v4u*)(Vt + (size_t)vc[1] * NVOX + nk0 + vkc[1] * 8);
        v4u vr2 = *(const v4u*)(Vt + (size_t)vc[2] * NVOX + nk0 + vkc[2] * 8);
        v4u vr3 = *(const v4u*)(Vt + (size_t)vc[3] * NVOX + nk0 + vkc[3] * 8);
#endif

        // ---- scores ----
        const v8f zero8 = {0,0,0,0,0,0,0,0};
        v8f S0 = __builtin_amdgcn_wmma_f32_16x16x32_f16(false, aq.v, false, bk0.v,
                                                        (short)0, zero8, false, false);
        v8f S1 = __builtin_amdgcn_wmma_f32_16x16x32_f16(false, aq.v, false, bk1.v,
                                                        (short)0, zero8, false, false);

        // ---- prefetch next K fragments ----
        if (lane < 16) {
            const v4u* kp0 = (const v4u*)(Kh + (size_t)(nk0 + lane) * RCH);
            bk0.q[0] = kp0[0]; bk0.q[1] = kp0[1];
            const v4u* kp1 = (const v4u*)(Kh + (size_t)(nk0 + 16 + lane) * RCH);
            bk1.q[0] = kp1[0]; bk1.q[1] = kp1[1];
        }

        // ---- online softmax with tile-wide max (log2 domain) ----
        float t = fmaxf(S0[0], S1[0]);
        #pragma unroll
        for (int j = 1; j < 8; ++j) t = fmaxf(t, fmaxf(S0[j], S1[j]));
        t = fmaxf(t, __shfl_xor(t, 1, 32));
        t = fmaxf(t, __shfl_xor(t, 2, 32));
        t = fmaxf(t, __shfl_xor(t, 4, 32));
        t = fmaxf(t, __shfl_xor(t, 8, 32));
        t = fmaxf(t, __shfl_xor(t, 16, 32));
        const float mnew  = fmaxf(m, t);
        const float alpha = exp2f(m - mnew);
        m = mnew;

        #pragma unroll
        for (int ct = 0; ct < 9; ++ct)
            #pragma unroll
            for (int j = 0; j < 8; ++j) O[ct][j] *= alpha;

        #pragma unroll
        for (int j = 0; j < 8; ++j) {
            float p0 = exp2f(S0[j] - mnew);
            float p1 = exp2f(S1[j] - mnew);
            int row = hi * 8 + j;
            pbuf[row * PPITCH + l16]      = (_Float16)p0;
            pbuf[row * PPITCH + 16 + l16] = (_Float16)p1;
        }

        asm volatile("s_wait_dscnt 0x0" ::: "memory");   // own-wave P stores -> loads

        FragH ap;   // P as A-matrix 16x32 f16 fragment (transposed via LDS)
        {
            const v4u* pp = (const v4u*)(pbuf + l16 * PPITCH);
            ap.q[0] = pp[2 * hi];        // keys 0..7 | 16..23
            ap.q[1] = pp[2 * hi + 1];    // keys 8..15 | 24..31
        }

        // ---- O += P x V over 8 channel tiles, plus row-sum column ----
        const _Float16* vb = ldsV[cur];
        #pragma unroll
        for (int ct = 0; ct < 8; ++ct) {
            FragH bv;
            const v4u* vp = (const v4u*)(vb + (ct * 16 + l16) * VPITCH + hi * 16);
            bv.q[0] = vp[0]; bv.q[1] = vp[1];
            O[ct] = __builtin_amdgcn_wmma_f32_16x16x32_f16(false, ap.v, false, bv.v,
                                                           (short)0, O[ct], false, false);
        }
        O[8] = __builtin_amdgcn_wmma_f32_16x16x32_f16(false, ap.v, false, ones.v,
                                                      (short)0, O[8], false, false);

#if !USE_ASYNC
        // ---- commit prefetched V tile to the other buffer ----
        _Float16* wb = ldsV[cur ^ 1];
        *(v4u*)(&wb[vc[0] * VPITCH + vkc[0] * 8]) = vr0;
        *(v4u*)(&wb[vc[1] * VPITCH + vkc[1] * 8]) = vr1;
        *(v4u*)(&wb[vc[2] * VPITCH + vkc[2] * 8]) = vr2;
        *(v4u*)(&wb[vc[3] * VPITCH + vkc[3] * 8]) = vr3;
#endif

        cur ^= 1;
    }

    // ---- epilogue: out = x_q + gamma * (O / rowsum), rowsum lives in O[8] ----
    const float g = gamma[0];
    #pragma unroll
    for (int j = 0; j < 8; ++j) {
        float inv = 1.0f / O[8][j];
        int n = qbase + hi * 8 + j;
        #pragma unroll
        for (int ct = 0; ct < 8; ++ct) {
            int c = ct * 16 + l16;
            size_t off = (size_t)c * NVOX + n;
            out[off] = xq[off] + g * (O[ct][j] * inv);
        }
    }
}

extern "C" void kernel_launch(void* const* d_in, const int* in_sizes, int n_in,
                              void* d_out, int out_size, void* d_ws, size_t ws_size,
                              hipStream_t stream)
{
    const float* xq    = (const float*)d_in[0];
    const float* xkv   = (const float*)d_in[1];
    const float* Wq    = (const float*)d_in[2];
    const float* bq    = (const float*)d_in[3];
    const float* Wk    = (const float*)d_in[4];
    const float* bk    = (const float*)d_in[5];
    const float* Wv    = (const float*)d_in[6];
    const float* bv    = (const float*)d_in[7];
    const float* gamma = (const float*)d_in[8];
    float* out = (float*)d_out;

    _Float16* Qh = (_Float16*)d_ws;                  // N*16 f16
    _Float16* Kh = Qh + (size_t)NVOX * RCH;          // N*16 f16
    _Float16* Vt = Kh + (size_t)NVOX * RCH;          // C*N f16

    ca_proj_qk<<<NVOX / 8, 256, 0, stream>>>(xq, xkv, Wq, bq, Wk, bk, Qh, Kh);
    ca_proj_v<<<(CCH * NVOX) / 256, 256, 0, stream>>>(xkv, Wv, bv, Vt);
    ca_flash_attn<<<NVOX / 64, 128, 0, stream>>>(Qh, Kh, Vt, xq, gamma, out);
}
